// Invoice_BiLSTM_CNN_CRF_13400297963959
// MI455X (gfx1250) — matrix operand order repaired and checked by reading, not compiled, not run on hardware
//
#include <hip/hip_runtime.h>
#include <hip/hip_bf16.h>

// ---------------- problem dimensions ----------------
#define Bsz   32
#define Ssz   512
#define Din   330
#define DP    352          // Din padded to multiple of 32 (K-tile)
#define Hn    256
#define Gn    1024         // 4*H gate columns
#define H2    512          // 2*H concat
#define KC    10           // CRF tags
#define KP    16           // padded tag dim for WMMA
#define Mrows (Ssz*Bsz)    // 16384 time-major rows

#define NSLICE 16          // H-slices per direction in the recurrent kernel
#define HS     16          // h units per slice
#define NCOLS  64          // 4 gates * HS columns per slice

typedef __attribute__((ext_vector_type(16))) __bf16 v16bf;
typedef __attribute__((ext_vector_type(8)))  float  v8f;
typedef __attribute__((ext_vector_type(4)))  int    v4i;

union FragAB { v16bf v; uint4 q[2]; unsigned short u[16]; };

// ---------------- CDNA5 async copy to LDS (probe-guarded) ----------------
#if defined(__gfx1250__) && __has_builtin(__builtin_amdgcn_global_load_async_to_lds_b128)
#define HAVE_ASYNC 1
#else
#define HAVE_ASYNC 0
#endif

// copy 16 bytes global -> LDS through the async data mover (ASYNCcnt)
__device__ __forceinline__ void async_copy16(const void* gsrc, void* ldst) {
#if HAVE_ASYNC
    __builtin_amdgcn_global_load_async_to_lds_b128(
        (__attribute__((address_space(1))) v4i*)gsrc,
        (__attribute__((address_space(3))) v4i*)ldst, 0, 0);
#else
    *(uint4*)ldst = *(const uint4*)gsrc;
#endif
}

__device__ __forceinline__ void async_wait_all() {
#if HAVE_ASYNC
#if __has_builtin(__builtin_amdgcn_s_wait_asynccnt)
    __builtin_amdgcn_s_wait_asynccnt(0);
#else
    asm volatile("s_wait_asynccnt 0" ::: "memory");
#endif
#endif
}

__device__ inline unsigned short f2bf(float f) {
    union { float f; unsigned u; } v; v.f = f;
    unsigned r = v.u + 0x7FFFu + ((v.u >> 16) & 1u);   // round-to-nearest-even
    return (unsigned short)(r >> 16);
}

// ---------------- packing / init kernels ----------------

// x [B,S,D] f32 -> xT [S,B,DP] bf16 (zero padded)
__global__ void pack_x_kernel(const float* __restrict__ x, unsigned short* __restrict__ xT) {
    long i = (long)blockIdx.x * blockDim.x + threadIdx.x;
    const long total = (long)Mrows * DP;
    if (i >= total) return;
    int d = (int)(i % DP);
    long sb = i / DP;
    int b = (int)(sb % Bsz);
    int s = (int)(sb / Bsz);
    float val = (d < Din) ? x[((long)b * Ssz + s) * Din + d] : 0.f;
    xT[i] = f2bf(val);
}

// pack W_ih (both dirs, padded), W_hh (both dirs), W_fc (padded rows) to bf16
__global__ void pack_w_kernel(const float* __restrict__ Wih_f, const float* __restrict__ Whh_f,
                              const float* __restrict__ Wih_b, const float* __restrict__ Whh_b,
                              const float* __restrict__ Wfc,
                              unsigned short* __restrict__ wih,
                              unsigned short* __restrict__ whh,
                              unsigned short* __restrict__ wfc) {
    long i = (long)blockIdx.x * blockDim.x + threadIdx.x;
    const long nih = (long)2 * Gn * DP;
    const long nhh = (long)2 * Gn * Hn;
    const long nfc = (long)KP * H2;
    if (i < nih) {
        int d = (int)(i % DP);
        long gd = i / DP;
        int g = (int)(gd % Gn);
        int dir = (int)(gd / Gn);
        const float* W = dir ? Wih_b : Wih_f;
        wih[i] = f2bf(d < Din ? W[(long)g * Din + d] : 0.f);
        return;
    }
    i -= nih;
    if (i < nhh) {
        int k = (int)(i % Hn);
        long gk = i / Hn;
        int g = (int)(gk % Gn);
        int dir = (int)(gk / Gn);
        const float* W = dir ? Whh_b : Whh_f;
        whh[i] = f2bf(W[(long)g * Hn + k]);
        return;
    }
    i -= nhh;
    if (i < nfc) {
        int k = (int)(i % H2);
        int n = (int)(i / H2);
        wfc[i] = f2bf(n < KC ? Wfc[(long)n * H2 + k] : 0.f);
    }
}

// zero h double-buffers and the grid-barrier counter (re-run every launch)
__global__ void init_state_kernel(unsigned short* __restrict__ hbuf, int* __restrict__ barrier_cnt) {
    long i = (long)blockIdx.x * blockDim.x + threadIdx.x;
    const long total = (long)2 * 2 * Bsz * Hn;   // dirs * dbl-buffer * B * H
    if (i < total) hbuf[i] = 0;
    if (i == 0) *barrier_cnt = 0;
}

// ---------------- kernel 1: input projection GEMM (WMMA, async-staged A) ----------------
// pre[dir][s*B+b][g] = xT @ W_ih^T + bias   M=16384 N=1024 K=352
// All 4 waves in a block share one 16-row A strip -> stage it in LDS via async copy.
__global__ void input_proj_kernel(const unsigned short* __restrict__ xT,
                                  const unsigned short* __restrict__ wih,
                                  const float* __restrict__ b_f, const float* __restrict__ b_b,
                                  float* __restrict__ pre) {
    __shared__ unsigned short a_s[16][DP];      // 11 KB shared A strip
    const int tid  = threadIdx.x;
    const int wave = tid >> 5;                  // 4 waves / block
    const int lane = tid & 31;
    const int half = lane >> 4, l16 = lane & 15;
    const int mt = blockIdx.y;                  // 0..1023
    const int nt = blockIdx.x * 4 + wave;       // 0..63
    const int dir = blockIdx.z;

    // async stage A strip (16 x DP bf16) into LDS: 704 16-byte chunks
    for (int idx = tid; idx < 16 * (DP / 8); idx += blockDim.x) {
        const int row = idx / (DP / 8);
        const int c8  = idx % (DP / 8);
        async_copy16(xT + ((long)(mt * 16 + row)) * DP + c8 * 8, &a_s[row][c8 * 8]);
    }
    async_wait_all();
    __syncthreads();

    const unsigned short* W = wih + (long)dir * Gn * DP;
    const float* bias = dir ? b_b : b_f;
    const int ncol = nt * 16 + l16;             // B col (N) for this lane
    v8f acc = {};
    for (int kt = 0; kt < DP / 32; ++kt) {
        const int k0 = kt * 32;
        FragAB a, b;
        const unsigned short* ap = &a_s[l16][k0 + half * 16];
        a.q[0] = *(const uint4*)(ap);           // ds_load_b128: k = k0 + half*16 .. +7
        a.q[1] = *(const uint4*)(ap + 8);       //               k = k0 + half*16 + 8 .. +15
        const unsigned short* bp = W + (long)ncol * DP + k0 + half * 16;
        b.q[0] = *(const uint4*)(bp);
        b.q[1] = *(const uint4*)(bp + 8);
        acc = __builtin_amdgcn_wmma_f32_16x16x32_bf16(false, a.v, false, b.v,
                                                      (short)0, acc, false, false);
    }
    const float bv = bias[ncol];
    float* out = pre + (long)dir * Mrows * Gn;
    #pragma unroll
    for (int r = 0; r < 8; ++r) {
        const int m = mt * 16 + r + 8 * half;   // C/D layout: VGPR r -> M = r + 8*half
        out[(long)m * Gn + ncol] = acc[r] + bv;
    }
}

// ---------------- kernel 2: persistent BiLSTM recurrence (WMMA + async LDS) ----------------
// grid = 32 WGs: blockIdx [0..15] = forward slices, [16..31] = backward slices.
// Each WG owns HS=16 h-units: its 64 gate columns of W_hh live in LDS for all 512 steps.
__global__ void lstm_scan_kernel(const float* __restrict__ pre,
                                 const unsigned short* __restrict__ whh,
                                 unsigned short* __restrict__ hbuf,     // [dir][2][B][H] bf16
                                 unsigned short* __restrict__ hcat,     // [S][B][H2] bf16
                                 int* __restrict__ barrier_cnt) {
    __shared__ unsigned short whh_s[NCOLS][Hn];   // 32 KB: B-operand, WMMA-fed from LDS
    __shared__ unsigned short hprev_s[Bsz][Hn];   // 16 KB: per-step A operand (async staged)
    __shared__ float gbuf[4][Bsz][HS];            //  8 KB: gate exchange between waves
    __shared__ float cst[Bsz][HS];                //  2 KB: persistent cell state

    const int dir   = blockIdx.x >> 4;
    const int slice = blockIdx.x & 15;
    const int tid   = threadIdx.x;
    const int wave  = tid >> 5, lane = tid & 31;
    const int half  = lane >> 4, l16 = lane & 15;

    // async-stage this slice's W_hh columns into LDS: local col c = gate*HS + u
    const unsigned short* Wd = whh + (long)dir * Gn * Hn;
    for (int idx = tid; idx < NCOLS * (Hn / 8); idx += blockDim.x) {
        const int k8 = idx % (Hn / 8);
        const int c  = idx / (Hn / 8);
        const int gate = c >> 4, u = c & 15;
        const long gcol = (long)gate * Hn + slice * HS + u;
        async_copy16(&Wd[gcol * Hn + k8 * 8], &whh_s[c][k8 * 8]);
    }
    for (int idx = tid; idx < Bsz * HS; idx += blockDim.x) ((float*)cst)[idx] = 0.f;
    async_wait_all();
    __syncthreads();

    const float* pre_d = pre + (long)dir * Mrows * Gn;
    unsigned short* hb = hbuf + (long)dir * 2 * Bsz * Hn;

    // wave -> output tile: 2 M-tiles x 4 N-tiles (one N-tile == one gate)
    const int mt     = wave & 1;
    const int gate_w = wave >> 1;
    const int ncol_l = gate_w * 16 + l16;                       // local B col
    const long gcol  = (long)gate_w * Hn + slice * HS + l16;    // global gate col
    const int  brow  = mt * 16 + l16;                           // A row = batch idx

    for (int t = 0; t < Ssz; ++t) {
        const int time = dir ? (Ssz - 1 - t) : t;
        const unsigned short* hprev = hb + (long)(t & 1) * Bsz * Hn;
        unsigned short*       hnext = hb + (long)((t + 1) & 1) * Bsz * Hn;

        // async-stage h_{t-1} (32x256 bf16) into LDS: 1024 16-byte chunks
        for (int idx = tid; idx < Bsz * (Hn / 8); idx += blockDim.x) {
            const int row = idx / (Hn / 8);
            const int c8  = idx % (Hn / 8);
            async_copy16(hprev + (long)row * Hn + c8 * 8, &hprev_s[row][c8 * 8]);
        }
        // prefetch next step's pre rows while the async copy + WMMAs run
        if (t + 1 < Ssz) {
            const int tn = dir ? (Ssz - 2 - t) : (t + 1);
            __builtin_prefetch(&pre_d[((long)tn * Bsz + brow) * Gn + gcol], 0, 1);
        }
        async_wait_all();
        __syncthreads();

        v8f acc = {};
        #pragma unroll
        for (int kt = 0; kt < Hn / 32; ++kt) {
            const int k0 = kt * 32;
            FragAB a, b;
            const unsigned short* ap = &hprev_s[brow][k0 + half * 16];
            a.q[0] = *(const uint4*)(ap);                        // ds_load_b128
            a.q[1] = *(const uint4*)(ap + 8);
            const unsigned short* bp2 = &whh_s[ncol_l][k0 + half * 16];
            b.q[0] = *(const uint4*)(bp2);                       // ds_load_b128
            b.q[1] = *(const uint4*)(bp2 + 8);
            acc = __builtin_amdgcn_wmma_f32_16x16x32_bf16(false, a.v, false, b.v,
                                                          (short)0, acc, false, false);
        }
        #pragma unroll
        for (int r = 0; r < 8; ++r) {
            const int bidx = mt * 16 + r + 8 * half;
            gbuf[gate_w][bidx][l16] = acc[r] + pre_d[((long)time * Bsz + bidx) * Gn + gcol];
        }
        __syncthreads();

        // LSTM cell activations (i,f,g,o order)
        for (int idx = tid; idx < Bsz * HS; idx += blockDim.x) {
            const int u = idx & (HS - 1);
            const int bidx = idx >> 4;
            const float gi = gbuf[0][bidx][u], gf = gbuf[1][bidx][u];
            const float gg = gbuf[2][bidx][u], go = gbuf[3][bidx][u];
            const float si = 1.f / (1.f + __expf(-gi));
            const float sf = 1.f / (1.f + __expf(-gf));
            const float so = 1.f / (1.f + __expf(-go));
            const float c = sf * cst[bidx][u] + si * tanhf(gg);
            cst[bidx][u] = c;
            const unsigned short hbf = f2bf(so * tanhf(c));
            const int col = slice * HS + u;
            hnext[(long)bidx * Hn + col] = hbf;
            hcat[((long)time * Bsz + bidx) * H2 + dir * Hn + col] = hbf;
        }
        __syncthreads();

        // grid-wide step barrier across 32 persistent WGs (monotonic counter)
        __threadfence();
        if (tid == 0) {
            atomicAdd(barrier_cnt, 1);
            const int target = 32 * (t + 1);
            while (__hip_atomic_load(barrier_cnt, __ATOMIC_ACQUIRE,
                                     __HIP_MEMORY_SCOPE_AGENT) < target)
                __builtin_amdgcn_s_sleep(1);
        }
#if __has_builtin(__builtin_amdgcn_s_cluster_barrier)
        // cluster-dispatch sync point (s_barrier_signal/wait -3; NOP when not in a cluster)
        __builtin_amdgcn_s_cluster_barrier();
#endif
        __syncthreads();
    }
}

// ---------------- kernel 3: FC emissions (WMMA) ----------------
// em[s*B+b][0..15] = hcat @ W_fc^T + b_fc   M=16384 N=16 K=512
__global__ void fc_kernel(const unsigned short* __restrict__ hcat,
                          const unsigned short* __restrict__ wfc,
                          const float* __restrict__ b_fc,
                          float* __restrict__ em) {
    const int wave = threadIdx.x >> 5, lane = threadIdx.x & 31;
    const int half = lane >> 4, l16 = lane & 15;
    const int mt = blockIdx.x * 4 + wave;
    const int row = mt * 16 + l16;
    v8f acc = {};
    for (int kt = 0; kt < H2 / 32; ++kt) {
        const int k0 = kt * 32;
        FragAB a, b;
        const unsigned short* ap = hcat + (long)row * H2 + k0 + half * 16;
        a.q[0] = *(const uint4*)(ap);
        a.q[1] = *(const uint4*)(ap + 8);
        const unsigned short* bp = wfc + (long)l16 * H2 + k0 + half * 16;
        b.q[0] = *(const uint4*)(bp);
        b.q[1] = *(const uint4*)(bp + 8);
        acc = __builtin_amdgcn_wmma_f32_16x16x32_bf16(false, a.v, false, b.v,
                                                      (short)0, acc, false, false);
    }
    const float bv = (l16 < KC) ? b_fc[l16] : 0.f;
    #pragma unroll
    for (int r = 0; r < 8; ++r) {
        const int m = mt * 16 + r + 8 * half;
        em[(long)m * KP + l16] = acc[r] + bv;
    }
}

// ---------------- kernel 4: CRF NLL + Viterbi (one lane per batch element) ----------------
__global__ void crf_kernel(const float* __restrict__ em, const int* __restrict__ mask,
                           const int* __restrict__ labels,
                           const float* __restrict__ trans, const float* __restrict__ startt,
                           const float* __restrict__ endt,
                           int* __restrict__ bp, float* __restrict__ out) {
    __shared__ float red[Bsz];
    __shared__ float tr[KC * KC];
    const int b = threadIdx.x;
    for (int i = b; i < KC * KC; i += Bsz) tr[i] = trans[i];
    __syncthreads();

    // ---- forward algorithm + gold-path score ----
    float alpha[KC], nalpha[KC];
    int tag0 = labels[b * Ssz + 0];
    float score = startt[tag0] + em[(long)b * KP + tag0];
    int prev = tag0;
    for (int k = 0; k < KC; ++k) alpha[k] = startt[k] + em[(long)b * KP + k];

    for (int s = 1; s < Ssz; ++s) {
        const float* e = em + ((long)s * Bsz + b) * KP;
        const float m = (float)mask[b * Ssz + s];
        const int tg = labels[b * Ssz + s];
        score += (tr[prev * KC + tg] + e[tg]) * m;
        if (m > 0.f) prev = tg;
        for (int j = 0; j < KC; ++j) {
            float mx = -1e30f;
            for (int k = 0; k < KC; ++k) mx = fmaxf(mx, alpha[k] + tr[k * KC + j]);
            float ssum = 0.f;
            for (int k = 0; k < KC; ++k) ssum += __expf(alpha[k] + tr[k * KC + j] - mx);
            nalpha[j] = mx + __logf(ssum) + e[j];
        }
        for (int j = 0; j < KC; ++j) alpha[j] = (m > 0.f) ? nalpha[j] : alpha[j];
    }
    score += endt[prev];
    float mx = -1e30f;
    for (int k = 0; k < KC; ++k) mx = fmaxf(mx, alpha[k] + endt[k]);
    float ssum = 0.f;
    for (int k = 0; k < KC; ++k) ssum += __expf(alpha[k] + endt[k] - mx);
    red[b] = (mx + __logf(ssum)) - score;   // per-sample NLL
    __syncthreads();
    if (b == 0) {
        float tot = 0.f;
        for (int i = 0; i < Bsz; ++i) tot += red[i];
        out[0] = tot / (float)Bsz;
    }

    // ---- Viterbi ----
    float vit[KC], nv[KC];
    for (int k = 0; k < KC; ++k) vit[k] = startt[k] + em[(long)b * KP + k];
    for (int s = 1; s < Ssz; ++s) {
        const float* e = em + ((long)s * Bsz + b) * KP;
        const float m = (float)mask[b * Ssz + s];
        for (int j = 0; j < KC; ++j) {
            float best = -1e30f; int bi = 0;
            for (int k = 0; k < KC; ++k) {
                const float v = vit[k] + tr[k * KC + j];
                if (v > best) { best = v; bi = k; }
            }
            nv[j] = best + e[j];
            bp[((long)s * Bsz + b) * KC + j] = bi;
        }
        for (int j = 0; j < KC; ++j) vit[j] = (m > 0.f) ? nv[j] : vit[j];
    }
    float best = -1e30f; int tag = 0;
    for (int k = 0; k < KC; ++k) {
        const float v = vit[k] + endt[k];
        if (v > best) { best = v; tag = k; }
    }
    out[1 + (long)b * Ssz + (Ssz - 1)] = (float)(mask[b * Ssz + Ssz - 1] > 0 ? tag : 0);
    for (int s = Ssz - 2; s >= 0; --s) {
        const int m = mask[b * Ssz + s + 1];
        const int nt2 = bp[((long)(s + 1) * Bsz + b) * KC + tag];
        tag = (m > 0) ? nt2 : tag;
        out[1 + (long)b * Ssz + s] = (float)(mask[b * Ssz + s] > 0 ? tag : 0);
    }
}

// ---------------- host launcher ----------------
extern "C" void kernel_launch(void* const* d_in, const int* in_sizes, int n_in,
                              void* d_out, int out_size, void* d_ws, size_t ws_size,
                              hipStream_t stream) {
    const float* x      = (const float*)d_in[0];
    const int*   mask   = (const int*)d_in[1];
    const int*   labels = (const int*)d_in[2];
    const float* Wih_f  = (const float*)d_in[3];
    const float* Whh_f  = (const float*)d_in[4];
    const float* b_f    = (const float*)d_in[5];
    const float* Wih_b  = (const float*)d_in[6];
    const float* Whh_b  = (const float*)d_in[7];
    const float* b_b    = (const float*)d_in[8];
    const float* Wfc    = (const float*)d_in[9];
    const float* b_fc   = (const float*)d_in[10];
    const float* trans  = (const float*)d_in[11];
    const float* st     = (const float*)d_in[12];
    const float* en     = (const float*)d_in[13];
    float* out = (float*)d_out;

    char* ws = (char*)d_ws;
    size_t off = 0;
    auto alloc = [&](size_t bytes) {
        size_t p = off;
        off = (off + bytes + 255) & ~(size_t)255;
        return p;
    };
    unsigned short* xT      = (unsigned short*)(ws + alloc((size_t)Mrows * DP * 2));
    unsigned short* wih     = (unsigned short*)(ws + alloc((size_t)2 * Gn * DP * 2));
    unsigned short* whh     = (unsigned short*)(ws + alloc((size_t)2 * Gn * Hn * 2));
    unsigned short* wfc     = (unsigned short*)(ws + alloc((size_t)KP * H2 * 2));
    float*          pre     = (float*)(ws + alloc((size_t)2 * Mrows * Gn * 4));
    unsigned short* hbuf    = (unsigned short*)(ws + alloc((size_t)2 * 2 * Bsz * Hn * 2));
    unsigned short* hcat    = (unsigned short*)(ws + alloc((size_t)Mrows * H2 * 2));
    float*          em      = (float*)(ws + alloc((size_t)Mrows * KP * 4));
    int*            bpbuf   = (int*)(ws + alloc((size_t)Ssz * Bsz * KC * 4));
    int*            barrier = (int*)(ws + alloc(256));

    // stage 0: pack / init
    {
        long total = (long)Mrows * DP;
        pack_x_kernel<<<(unsigned)((total + 255) / 256), 256, 0, stream>>>(x, xT);
    }
    {
        long total = (long)2 * Gn * DP + (long)2 * Gn * Hn + (long)KP * H2;
        pack_w_kernel<<<(unsigned)((total + 255) / 256), 256, 0, stream>>>(
            Wih_f, Whh_f, Wih_b, Whh_b, Wfc, wih, whh, wfc);
    }
    {
        long total = (long)2 * 2 * Bsz * Hn;
        init_state_kernel<<<(unsigned)((total + 255) / 256), 256, 0, stream>>>(hbuf, barrier);
    }
    // stage 1: input projection GEMM (WMMA + async-staged LDS A-tile)
    input_proj_kernel<<<dim3(16, 1024, 2), 128, 0, stream>>>(xT, wih, b_f, b_b, pre);
    // stage 2: persistent BiLSTM scan (WMMA + async LDS staging + grid barrier)
    lstm_scan_kernel<<<32, 256, 0, stream>>>(pre, whh, hbuf, hcat, barrier);
    // stage 3: FC emissions (WMMA)
    fc_kernel<<<256, 128, 0, stream>>>(hcat, wfc, b_fc, em);
    // stage 4: CRF loss + Viterbi decode
    crf_kernel<<<1, Bsz, 0, stream>>>(em, mask, labels, trans, st, en, bpbuf, out);
    (void)in_sizes; (void)n_in; (void)out_size; (void)ws_size;
}
